// SelfAttention_3332894622709
// MI455X (gfx1250) — matrix operand order repaired and checked by reading, not compiled, not run on hardware
//
#include <hip/hip_runtime.h>
#include <hip/hip_bf16.h>

// ---- CDNA5 wave32 WMMA types ----
typedef __attribute__((ext_vector_type(16))) _Float16 v16h;
typedef __attribute__((ext_vector_type(8)))  _Float16 v8h;
typedef __attribute__((ext_vector_type(8)))  float    v8f;

static __device__ __forceinline__ v16h pack16(v8h lo, v8h hi) {
    v16h r;
#pragma unroll
    for (int i = 0; i < 8; ++i) { r[i] = lo[i]; r[i + 8] = hi[i]; }
    return r;
}

static __device__ __forceinline__ v8f wmma32(v16h a, v16h b, v8f c) {
    // D = A(16x32 f16) * B(32x16 f16) + C(16x16 f32)
    return __builtin_amdgcn_wmma_f32_16x16x32_f16(
        /*neg_a=*/false, a, /*neg_b=*/false, b,
        /*c_mod=*/(short)0, c, /*reuse_a=*/false, /*reuse_b=*/false);
}

// A-operand load: 16x32 f16 tile, lane M = lane%16, half-group g splits K
static __device__ __forceinline__ v16h loadA(const _Float16* row, int k0, int g) {
    return *(const v16h*)(row + k0 + 16 * g);
}

// Problem constants
static constexpr int B_ = 16, S_ = 1024, F_ = 512, H_ = 8, DK_ = 64;
static constexpr int M_ = B_ * S_;           // 16384 rows
static constexpr int N_ = H_ * DK_;          // 512
static constexpr int KD_ = 512;              // K depth of both GEMMs (== ld)

// ---------------------------------------------------------------------------
// f32 -> f16 conversion, 8 elements/thread (b128 in, b128 out)
// ---------------------------------------------------------------------------
__global__ void cvt_f32_f16(const float* __restrict__ in,
                            _Float16* __restrict__ out, int n8) {
    int i = blockIdx.x * blockDim.x + threadIdx.x;
    const int stride = gridDim.x * blockDim.x;
    for (; i < n8; i += stride) {
        const float4* p = (const float4*)(in + (size_t)i * 8);
        const float4 a = p[0], b = p[1];
        v8h o;
        o[0] = (_Float16)a.x; o[1] = (_Float16)a.y;
        o[2] = (_Float16)a.z; o[3] = (_Float16)a.w;
        o[4] = (_Float16)b.x; o[5] = (_Float16)b.y;
        o[6] = (_Float16)b.z; o[7] = (_Float16)b.w;
        *(v8h*)(out + (size_t)i * 8) = o;
    }
}

// ---------------------------------------------------------------------------
// Shared 32x64-tile GEMM core, K = 512, both matrices ld = 512.
// Ping-pong double buffering (unroll-by-2) => no buffer-rotate register moves.
// ---------------------------------------------------------------------------
static __device__ __forceinline__ void gemm_loadstep(
        const _Float16* xr0, const _Float16* xr1, const _Float16* wp,
        int k, int g, v16h& A0, v16h& A1, v16h* B) {
    A0 = loadA(xr0, k, g);
    A1 = loadA(xr1, k, g);
#pragma unroll
    for (int j = 0; j < 4; ++j)
        B[j] = *(const v16h*)(wp + k + (size_t)(j * 16) * KD_);
}

static __device__ __forceinline__ void gemm_mma(
        v8f* acc, const v16h& A0, const v16h& A1, const v16h* B) {
#pragma unroll
    for (int j = 0; j < 4; ++j) acc[j]     = wmma32(A0, B[j], acc[j]);
#pragma unroll
    for (int j = 0; j < 4; ++j) acc[4 + j] = wmma32(A1, B[j], acc[4 + j]);
}

static __device__ __forceinline__ void gemm_core(
        const _Float16* xr0, const _Float16* xr1, const _Float16* wp,
        int g, v8f* acc) {
    v16h A0a, A1a, Ba[4];
    v16h A0b, A1b, Bb[4];
    gemm_loadstep(xr0, xr1, wp, 0, g, A0a, A1a, Ba);
    for (int k0 = 0; k0 < KD_ - 64; k0 += 64) {
        gemm_loadstep(xr0, xr1, wp, k0 + 32, g, A0b, A1b, Bb);
        __builtin_prefetch(wp + k0 + 192, 0, 1);   // global_prefetch_b8
        gemm_mma(acc, A0a, A1a, Ba);
        gemm_loadstep(xr0, xr1, wp, k0 + 64, g, A0a, A1a, Ba);
        gemm_mma(acc, A0b, A1b, Bb);
    }
    gemm_loadstep(xr0, xr1, wp, KD_ - 32, g, A0b, A1b, Bb);
    gemm_mma(acc, A0a, A1a, Ba);     // k = KD_-64
    gemm_mma(acc, A0b, A1b, Bb);     // k = KD_-32
}

// ---------------------------------------------------------------------------
// QKV projection:  out[n, m] = sum_f X[m,f]*W[n,f] + bias[n]
// transposed==0 -> store [B,H,S,DK] ; transposed==1 -> store [B,H,DK,S]
// ---------------------------------------------------------------------------
__global__ __launch_bounds__(32)
void proj_gemm(const _Float16* __restrict__ X, const _Float16* __restrict__ W,
               const float* __restrict__ bias, _Float16* __restrict__ out,
               int transposed) {
    const int lane = threadIdx.x;
    const int g = lane >> 4;
    const int col = lane & 15;
    const int tileN = blockIdx.x & 7;        // 8 tiles of 64 cols
    const int tileM = blockIdx.x >> 3;       // 512 tiles of 32 rows
    const int m0 = tileM * 32;
    const int n0 = tileN * 64;

    const _Float16* xr0 = X + (size_t)(m0 + col) * F_;
    const _Float16* xr1 = X + (size_t)(m0 + 16 + col) * F_;
    const _Float16* wp  = W + (size_t)(n0 + col) * F_ + 16 * g;

    v8f acc[8] = {};   // acc[i*4+j]: i = M-half, j = N-16-chunk
    gemm_core(xr0, xr1, wp, g, acc);

#pragma unroll
    for (int i = 0; i < 2; ++i) {
#pragma unroll
        for (int j = 0; j < 4; ++j) {
            const int n = n0 + j * 16 + col;        // C/D: N = lane%16
            const float bv = bias[n];
            const int hh = n >> 6, d = n & 63;
#pragma unroll
            for (int r = 0; r < 8; ++r) {
                const int m = m0 + i * 16 + r + 8 * g;  // C/D: M = r + 8g
                const int bb = m >> 10, s = m & 1023;
                const float val = acc[i * 4 + j][r] + bv;
                size_t idx = transposed
                    ? (((size_t)(bb * H_ + hh) * DK_ + d) * S_ + s)
                    : (((size_t)(bb * H_ + hh) * S_ + s) * DK_ + d);
                out[idx] = (_Float16)val;
            }
        }
    }
}

// ---------------------------------------------------------------------------
// Output projection: out[m,f] = sum_n Z[m,n]*WO[f,n] + WOb[f]   (f32 out)
// ---------------------------------------------------------------------------
__global__ __launch_bounds__(32)
void out_gemm(const _Float16* __restrict__ Zm, const _Float16* __restrict__ W,
              const float* __restrict__ bias, float* __restrict__ out) {
    const int lane = threadIdx.x;
    const int g = lane >> 4;
    const int col = lane & 15;
    const int tileN = blockIdx.x & 7;
    const int tileM = blockIdx.x >> 3;
    const int m0 = tileM * 32;
    const int n0 = tileN * 64;

    const _Float16* zr0 = Zm + (size_t)(m0 + col) * N_;
    const _Float16* zr1 = Zm + (size_t)(m0 + 16 + col) * N_;
    const _Float16* wp  = W + (size_t)(n0 + col) * N_ + 16 * g;

    v8f acc[8] = {};
    gemm_core(zr0, zr1, wp, g, acc);

#pragma unroll
    for (int i = 0; i < 2; ++i) {
#pragma unroll
        for (int j = 0; j < 4; ++j) {
            const int f = n0 + j * 16 + col;
            const float bv = bias[f];
#pragma unroll
            for (int r = 0; r < 8; ++r) {
                const int m = m0 + i * 16 + r + 8 * g;
                out[(size_t)m * F_ + f] = acc[i * 4 + j][r] + bv;
            }
        }
    }
}

// ---------------------------------------------------------------------------
// Flash attention helpers
// ---------------------------------------------------------------------------
static __device__ __forceinline__ void loadKT(
        const _Float16* kbase, int t, int col, int g,
        v16h& k00, v16h& k01, v16h& k10, v16h& k11) {
    const _Float16* kr0 = kbase + (size_t)(t + col) * DK_ + 16 * g;
    const _Float16* kr1 = kbase + (size_t)(t + 16 + col) * DK_ + 16 * g;
    k00 = *(const v16h*)kr0;  k01 = *(const v16h*)(kr0 + 32);
    k10 = *(const v16h*)kr1;  k11 = *(const v16h*)(kr1 + 32);
}

// One 32-key step: scores -> mask/scale -> online softmax -> P bounce -> o +=
static __device__ __forceinline__ void attn_step(
        int t0, int limit, int col, int g,
        const v16h& a0, const v16h& a1,
        const v16h& k00, const v16h& k01, const v16h& k10, const v16h& k11,
        const _Float16* vbase, _Float16* pls,
        v8f& o0, v8f& o1, v8f& o2, v8f& o3, float* mrow, float* lrow) {
    // ---- V loads issue first: softmax VALU below hides their latency ----
    const _Float16* vr = vbase + (size_t)col * S_ + t0 + 16 * g;
    const v16h v0 = *(const v16h*)(vr);
    const v16h v1 = *(const v16h*)(vr + (size_t)16 * S_);
    const v16h v2 = *(const v16h*)(vr + (size_t)32 * S_);
    const v16h v3 = *(const v16h*)(vr + (size_t)48 * S_);

    // ---- scores for 32 keys: two 16x16 f32 tiles ----
    float f0v[8], f1v[8];
    {
        v8f sc = {};
        sc = wmma32(a0, k00, sc);
        sc = wmma32(a1, k01, sc);
        const float madd = (t0 + col >= limit) ? -999999.f : 0.f;
#pragma unroll
        for (int r = 0; r < 8; ++r) f0v[r] = sc[r] * 0.125f + madd;
    }
    {
        v8f sc = {};
        sc = wmma32(a0, k10, sc);
        sc = wmma32(a1, k11, sc);
        const float madd = (t0 + 16 + col >= limit) ? -999999.f : 0.f;
#pragma unroll
        for (int r = 0; r < 8; ++r) f1v[r] = sc[r] * 0.125f + madd;
    }

    // ---- online softmax (rows live across 16-lane half-groups) ----
    float pr0[8], pr1[8];
#pragma unroll
    for (int r = 0; r < 8; ++r) {
        float mx = fmaxf(f0v[r], f1v[r]);
        mx = fmaxf(mx, __shfl_xor(mx, 1, 32));
        mx = fmaxf(mx, __shfl_xor(mx, 2, 32));
        mx = fmaxf(mx, __shfl_xor(mx, 4, 32));
        mx = fmaxf(mx, __shfl_xor(mx, 8, 32));
        const float newm = fmaxf(mrow[r], mx);
        const float corr = __expf(mrow[r] - newm);
        mrow[r] = newm;
        const float p0 = __expf(f0v[r] - newm);
        const float p1 = __expf(f1v[r] - newm);
        float rs = p0 + p1;
        rs += __shfl_xor(rs, 1, 32);
        rs += __shfl_xor(rs, 2, 32);
        rs += __shfl_xor(rs, 4, 32);
        rs += __shfl_xor(rs, 8, 32);
        lrow[r] = lrow[r] * corr + rs;
        o0[r] *= corr; o1[r] *= corr; o2[r] *= corr; o3[r] *= corr;
        pr0[r] = p0; pr1[r] = p1;
    }

    // ---- P: C/D layout -> row-major f16 in LDS -> A layout ----
#pragma unroll
    for (int r = 0; r < 8; ++r) {
        pls[(r + 8 * g) * 32 + col]      = (_Float16)pr0[r];
        pls[(r + 8 * g) * 32 + 16 + col] = (_Float16)pr1[r];
    }
    __syncthreads();
    const v16h pa = *(const v16h*)(&pls[col * 32 + 16 * g]);
    __syncthreads();

    // ---- z(16x64) += P(16x32) x V(32x64) ----
    o0 = wmma32(pa, v0, o0);
    o1 = wmma32(pa, v1, o1);
    o2 = wmma32(pa, v2, o2);
    o3 = wmma32(pa, v3, o3);
}

// ---------------------------------------------------------------------------
// Flash attention: one wave handles 16 query rows for one (b,h).
// Q,K layout [B,H,S,DK]; V layout [B,H,DK,S] (transposed); Z layout [B,S,H*DK]
// Ping-pong K double-buffering (unroll-by-2, no register rotation moves).
// ---------------------------------------------------------------------------
__global__ __launch_bounds__(32)
void attn_kernel(const _Float16* __restrict__ Q, const _Float16* __restrict__ K,
                 const _Float16* __restrict__ VT, const int* __restrict__ xlen,
                 _Float16* __restrict__ Z) {
    __shared__ _Float16 pls[16 * 32];   // P tile bounce buffer (C/D -> A layout)

    const int lane = threadIdx.x;
    const int g = lane >> 4;
    const int col = lane & 15;
    const int qt = blockIdx.x & 63;          // query tile within sequence
    const int bh = blockIdx.x >> 6;          // flattened (b,h)
    const int b = bh >> 3, h = bh & 7;
    const int s0 = qt * 16;
    const int limit = xlen[b] - 1;           // keys >= limit masked (off-by-one kept)

    // Q tile in A layout: two 16x32 chunks over DK
    const _Float16* qrow = Q + ((size_t)bh * S_ + s0 + col) * DK_;
    const v16h a0 = *(const v16h*)(qrow + 16 * g);
    const v16h a1 = *(const v16h*)(qrow + 32 + 16 * g);

    float mrow[8], lrow[8];
    v8f o0 = {}, o1 = {}, o2 = {}, o3 = {};
#pragma unroll
    for (int r = 0; r < 8; ++r) { mrow[r] = -1e30f; lrow[r] = 0.f; }

    const _Float16* kbase = K + (size_t)bh * S_ * DK_;
    const _Float16* vbase = VT + (size_t)bh * DK_ * S_;

    // ping-pong K tile buffers
    v16h ka00, ka01, ka10, ka11;
    v16h kb00, kb01, kb10, kb11;
    loadKT(kbase, 0, col, g, ka00, ka01, ka10, ka11);

    for (int t0 = 0; t0 < S_ - 64; t0 += 64) {
        loadKT(kbase, t0 + 32, col, g, kb00, kb01, kb10, kb11);
        attn_step(t0, limit, col, g, a0, a1, ka00, ka01, ka10, ka11,
                  vbase, pls, o0, o1, o2, o3, mrow, lrow);
        loadKT(kbase, t0 + 64, col, g, ka00, ka01, ka10, ka11);
        attn_step(t0 + 32, limit, col, g, a0, a1, kb00, kb01, kb10, kb11,
                  vbase, pls, o0, o1, o2, o3, mrow, lrow);
    }
    loadKT(kbase, S_ - 32, col, g, kb00, kb01, kb10, kb11);
    attn_step(S_ - 64, limit, col, g, a0, a1, ka00, ka01, ka10, ka11,
              vbase, pls, o0, o1, o2, o3, mrow, lrow);
    attn_step(S_ - 32, limit, col, g, a0, a1, kb00, kb01, kb10, kb11,
              vbase, pls, o0, o1, o2, o3, mrow, lrow);

    // ---- normalize and store to Z[B,S,H*DK] as f16 ----
#pragma unroll
    for (int r = 0; r < 8; ++r) {
        const float inv = 1.f / lrow[r];
        const int s = s0 + r + 8 * g;
        _Float16* zr = Z + ((size_t)b * S_ + s) * N_ + h * DK_;
        zr[col]      = (_Float16)(o0[r] * inv);
        zr[16 + col] = (_Float16)(o1[r] * inv);
        zr[32 + col] = (_Float16)(o2[r] * inv);
        zr[48 + col] = (_Float16)(o3[r] * inv);
    }
}

// ---------------------------------------------------------------------------
extern "C" void kernel_launch(void* const* d_in, const int* in_sizes, int n_in,
                              void* d_out, int out_size, void* d_ws, size_t ws_size,
                              hipStream_t stream) {
    const float* x    = (const float*)d_in[0];
    const int*   xlen = (const int*)d_in[1];
    const float* WQw  = (const float*)d_in[2];
    const float* WQb  = (const float*)d_in[3];
    const float* WKw  = (const float*)d_in[4];
    const float* WKb  = (const float*)d_in[5];
    const float* WVw  = (const float*)d_in[6];
    const float* WVb  = (const float*)d_in[7];
    const float* WOw  = (const float*)d_in[8];
    const float* WOb  = (const float*)d_in[9];
    float* out = (float*)d_out;

    char* ws = (char*)d_ws;
    const size_t SZ = (size_t)M_ * F_ * sizeof(_Float16);   // 16 MiB
    const size_t WZ = (size_t)N_ * F_ * sizeof(_Float16);   // 512 KiB
    _Float16* xh  = (_Float16*)(ws);
    _Float16* qb  = (_Float16*)(ws + 1 * SZ);   // [B,H,S,DK]
    _Float16* kb  = (_Float16*)(ws + 2 * SZ);   // [B,H,S,DK]
    _Float16* vT  = (_Float16*)(ws + 3 * SZ);   // [B,H,DK,S]
    _Float16* zb  = (_Float16*)(ws + 4 * SZ);   // [B,S,H*DK]
    _Float16* wqh = (_Float16*)(ws + 5 * SZ);
    _Float16* wkh = (_Float16*)(ws + 5 * SZ + 1 * WZ);
    _Float16* wvh = (_Float16*)(ws + 5 * SZ + 2 * WZ);
    _Float16* woh = (_Float16*)(ws + 5 * SZ + 3 * WZ);

    // 1) precision conversion (8 elems/thread, b128 traffic)
    cvt_f32_f16<<<2048, 256, 0, stream>>>(x,   xh,  (M_ * F_) / 8);
    cvt_f32_f16<<<128, 256, 0, stream>>>(WQw, wqh, (N_ * F_) / 8);
    cvt_f32_f16<<<128, 256, 0, stream>>>(WKw, wkh, (N_ * F_) / 8);
    cvt_f32_f16<<<128, 256, 0, stream>>>(WVw, wvh, (N_ * F_) / 8);
    cvt_f32_f16<<<128, 256, 0, stream>>>(WOw, woh, (N_ * F_) / 8);

    // 2) QKV projections (V stored transposed for contiguous P*V B-operands)
    const int gemmGrid = (M_ / 32) * (N_ / 64);   // 4096 waves
    proj_gemm<<<gemmGrid, 32, 0, stream>>>(xh, wqh, WQb, qb, 0);
    proj_gemm<<<gemmGrid, 32, 0, stream>>>(xh, wkh, WKb, kb, 0);
    proj_gemm<<<gemmGrid, 32, 0, stream>>>(xh, wvh, WVb, vT, 1);

    // 3) flash attention, one wave per (b,h,16 query rows)
    attn_kernel<<<B_ * H_ * (S_ / 16), 32, 0, stream>>>(qb, kb, vT, xlen, zb);

    // 4) output projection + bias (f32 result)
    out_gemm<<<(M_ / 32) * (F_ / 64), 32, 0, stream>>>(zb, woh, WOb, out);
}
